// SelfAttention_61581241090431
// MI455X (gfx1250) — matrix operand order repaired and checked by reading, not compiled, not run on hardware
//
#include <hip/hip_runtime.h>
#include <hip/hip_bf16.h>
#include <math.h>

typedef __attribute__((ext_vector_type(16))) _Float16 v16h;
typedef __attribute__((ext_vector_type(8)))  _Float16 v8h;
typedef __attribute__((ext_vector_type(8)))  float    v8f;

constexpr int Bn = 4;
constexpr int Cn = 192;
constexpr int Ln = 4096;
constexpr int Dn = 64;   // CO = head dim

// ---------------------------------------------------------------------------
// Cross-lane xor-16 exchange via v_permlanex16_b32 (VALU, no DS counter wait).
// Identity selects: lane i of each 16-lane row reads element i of the other row.
// ---------------------------------------------------------------------------
__device__ __forceinline__ int xor16_i(int v) {
    return __builtin_amdgcn_permlanex16(v, v, 0x76543210, 0xfedcba98, false, false);
}
__device__ __forceinline__ float xor16_f(float v) {
    return __builtin_bit_cast(float, xor16_i(__builtin_bit_cast(int, v)));
}
__device__ __forceinline__ unsigned xor16_u(unsigned v) {
    return (unsigned)xor16_i((int)v);
}

// ---------------------------------------------------------------------------
// Projection kernel: f,g,h,v1 = W{q,k,v,v1} @ x   (per batch)
//  Km (keys  = f) : f16  [b][l][d]   (row l contiguous in d)
//  Qm (query = g) : f16  [b][l][d]
//  Vm (value = h) : f16  [b][d][l]   (row d contiguous in l)
//  V1             : f32  [b][d][l]
// ---------------------------------------------------------------------------
__global__ __launch_bounds__(256) void proj_kernel(
    const float* __restrict__ x,
    const float* __restrict__ Wq, const float* __restrict__ Wk,
    const float* __restrict__ Wv, const float* __restrict__ Wv1,
    _Float16* __restrict__ Km, _Float16* __restrict__ Qm,
    _Float16* __restrict__ Vm, float* __restrict__ V1)
{
    __shared__ float xs[Cn * 16];          // 12 KB: x[c][li] for a 16-wide l tile
    const int tid   = threadIdx.x;
    const int blk   = blockIdx.x;          // Bn * (Ln/16) blocks
    const int b     = blk >> 8;
    const int ltile = (blk & 255) << 4;

    const float* xb = x + (size_t)b * Cn * Ln;
#pragma unroll
    for (int k = 0; k < 12; ++k) {
        int idx = tid + k * 256;           // 0..3071
        int c = idx >> 4, li = idx & 15;
        xs[idx] = xb[(size_t)c * Ln + ltile + li];
    }
    __syncthreads();

    const int o    = tid & 63;
    const int lset = tid >> 6;             // 0..3 ; thread covers l = lset + 4*j
    const float* wq = Wq  + o * Cn;
    const float* wk = Wk  + o * Cn;
    const float* wv = Wv  + o * Cn;
    const float* w1 = Wv1 + o * Cn;

    float aq[4] = {0,0,0,0}, ak[4] = {0,0,0,0}, av[4] = {0,0,0,0}, a1[4] = {0,0,0,0};
    for (int c = 0; c < Cn; ++c) {
        float q = wq[c], k = wk[c], v = wv[c], u = w1[c];
#pragma unroll
        for (int j = 0; j < 4; ++j) {
            float xv = xs[c * 16 + lset + j * 4];
            aq[j] += q * xv;  ak[j] += k * xv;
            av[j] += v * xv;  a1[j] += u * xv;
        }
    }
#pragma unroll
    for (int j = 0; j < 4; ++j) {
        int l = ltile + lset + j * 4;
        Km[((size_t)b * Ln + l) * Dn + o] = (_Float16)aq[j];   // keys (f)
        Qm[((size_t)b * Ln + l) * Dn + o] = (_Float16)ak[j];   // queries (g)
        Vm[((size_t)b * Dn + o) * Ln + l] = (_Float16)av[j];   // values (h)
        V1[((size_t)b * Dn + o) * Ln + l] = a1[j];
    }
}

// ---------------------------------------------------------------------------
// Flash attention: one wave per 16-query tile, online softmax over 4096 keys.
// ---------------------------------------------------------------------------
#define SHUF16(lov, hiv) __builtin_shufflevector((lov), (hiv), \
        0,1,2,3,4,5,6,7,8,9,10,11,12,13,14,15)
#define WMMA(a, bmat, c) __builtin_amdgcn_wmma_f32_16x16x32_f16( \
        false, (a), false, (bmat), (short)0, (c), false, false)

__global__ __launch_bounds__(128) void attn_kernel(
    const _Float16* __restrict__ Km, const _Float16* __restrict__ Qm,
    const _Float16* __restrict__ Vm, const float* __restrict__ V1,
    const float* __restrict__ gammaP, float* __restrict__ out)
{
    const int lane  = threadIdx.x & 31;
    const int wid   = blockIdx.x * 4 + (threadIdx.x >> 5);
    const int b     = wid >> 8;                 // 256 m-tiles per batch
    const int mbase = (wid & 255) << 4;
    const int n     = lane & 15;                // N column owned by this lane
    const bool lo   = lane < 16;
    const int koffA = lo ? 0 : 16;              // A-operand K split (contiguous)
    const int koffB = lo ? 0 : 16;              // B-operand K split

    // Q as B-operand (fixed per wave): lane n holds Q[mbase+n][koffB + i] per chunk
    const _Float16* Qrow = Qm + ((size_t)b * Ln + (mbase + n)) * Dn;
    const v16h qb0 = *(const v16h*)(Qrow + koffB);
    const v16h qb1 = *(const v16h*)(Qrow + 32 + koffB);

    const v8f zero8 = {0.f,0.f,0.f,0.f,0.f,0.f,0.f,0.f};
    v8f oacc[4];
    oacc[0] = oacc[1] = oacc[2] = oacc[3] = zero8;
    float m_run = -INFINITY, s_run = 0.0f;

    const _Float16* Kb = Km + (size_t)b * Ln * Dn;
    const _Float16* Vb = Vm + (size_t)b * Dn * Ln;

    for (int lb = 0; lb < Ln; lb += 32) {
        // ---- S = K_tile x Q  (two 16x16 tiles covering 32 keys, K-dim 64) ----
        v8f s0 = zero8, s1 = zero8;
        {
            const _Float16* kr = Kb + (size_t)(lb + n) * Dn;
            v16h ka = SHUF16(*(const v8h*)(kr + koffA),      *(const v8h*)(kr + 8 + koffA));
            s0 = WMMA(ka, qb0, s0);
            ka      = SHUF16(*(const v8h*)(kr + 32 + koffA), *(const v8h*)(kr + 40 + koffA));
            s0 = WMMA(ka, qb1, s0);
        }
        {
            const _Float16* kr = Kb + (size_t)(lb + 16 + n) * Dn;
            v16h ka = SHUF16(*(const v8h*)(kr + koffA),      *(const v8h*)(kr + 8 + koffA));
            s1 = WMMA(ka, qb0, s1);
            ka      = SHUF16(*(const v8h*)(kr + 32 + koffA), *(const v8h*)(kr + 40 + koffA));
            s1 = WMMA(ka, qb1, s1);
        }

        // ---- online softmax over the l (row/M) axis for each column n ----
        float tmax = s0[0];
#pragma unroll
        for (int r = 0; r < 8; ++r) { tmax = fmaxf(tmax, s0[r]); tmax = fmaxf(tmax, s1[r]); }
        tmax = fmaxf(tmax, xor16_f(tmax));          // lanes n and n+16 share column
        const float m_new = fmaxf(m_run, tmax);
        const float corr  = __expf(m_run - m_new);
        m_run = m_new;

        float p0[8], p1[8], psum = 0.f;
#pragma unroll
        for (int r = 0; r < 8; ++r) {
            p0[r] = __expf(s0[r] - m_new);
            p1[r] = __expf(s1[r] - m_new);
            psum += p0[r] + p1[r];
        }
        psum  += xor16_f(psum);
        s_run  = s_run * corr + psum;
#pragma unroll
        for (int ch = 0; ch < 4; ++ch)
#pragma unroll
            for (int r = 0; r < 8; ++r) oacc[ch][r] *= corr;

        // ---- repack P (D-layout f32) into B-operand (32x16 f16) ----
        // Convert to f16, pack (p0,p1) per row, single permlanex16 per row.
        // lane<16 needs K=0..15 (tile0 M=0..15); lane>=16 needs K=16..31 (tile1)
        v16h pb;
#pragma unroll
        for (int r = 0; r < 8; ++r) {
            _Float16 h0 = (_Float16)p0[r];
            _Float16 h1 = (_Float16)p1[r];
            unsigned own = (unsigned)__builtin_bit_cast(unsigned short, h0)
                         | ((unsigned)__builtin_bit_cast(unsigned short, h1) << 16);
            unsigned oth = xor16_u(own);            // partner's (p0,p1) pair
            _Float16 o0 = __builtin_bit_cast(_Float16, (unsigned short)(oth & 0xffffu));
            _Float16 o1 = __builtin_bit_cast(_Float16, (unsigned short)(oth >> 16));
            pb[r]     = lo ? h0 : o1;               // K = koffB + r
            pb[r + 8] = lo ? o0 : h1;               // K = koffB + 8 + r
        }

        // ---- O += V_chunk x P  (4 chunks of 16 output channels, K=32 keys) ----
#pragma unroll
        for (int ch = 0; ch < 4; ++ch) {
            const _Float16* vr = Vb + (size_t)(ch * 16 + n) * Ln + lb;
            v16h va = SHUF16(*(const v8h*)(vr + koffA), *(const v8h*)(vr + 8 + koffA));
            oacc[ch] = WMMA(va, pb, oacc[ch]);
        }
    }

    // ---- epilogue: out = gamma * O / sum + v1 ----
    const float scale = (*gammaP) / s_run;
    float*       ob  = out + (size_t)b * Dn * Ln + mbase + n;
    const float* v1b = V1  + (size_t)b * Dn * Ln + mbase + n;
#pragma unroll
    for (int ch = 0; ch < 4; ++ch)
#pragma unroll
        for (int r = 0; r < 8; ++r) {
            int og = ch * 16 + r + (lo ? 0 : 8);    // D-layout: M = r + 8*hi
            size_t idx = (size_t)og * Ln;
            ob[idx] = oacc[ch][r] * scale + v1b[idx];
        }
}

// ---------------------------------------------------------------------------
extern "C" void kernel_launch(void* const* d_in, const int* in_sizes, int n_in,
                              void* d_out, int out_size, void* d_ws, size_t ws_size,
                              hipStream_t stream) {
    const float* x     = (const float*)d_in[0];
    const float* Wq    = (const float*)d_in[1];
    const float* Wk    = (const float*)d_in[2];
    const float* Wv    = (const float*)d_in[3];
    const float* Wv1   = (const float*)d_in[4];
    const float* gamma = (const float*)d_in[5];
    float* out = (float*)d_out;

    // workspace: Km(2MB f16) | Qm(2MB f16) | Vm(2MB f16) | V1(4MB f32) = 10MB
    char* ws = (char*)d_ws;
    _Float16* Km = (_Float16*)(ws);
    _Float16* Qm = (_Float16*)(ws + (size_t)2 * 1024 * 1024);
    _Float16* Vm = (_Float16*)(ws + (size_t)4 * 1024 * 1024);
    float*    V1 = (float*)   (ws + (size_t)6 * 1024 * 1024);

    proj_kernel<<<Bn * (Ln / 16), 256, 0, stream>>>(x, Wq, Wk, Wv, Wv1, Km, Qm, Vm, V1);
    attn_kernel<<<(Bn * (Ln / 16)) / 4, 128, 0, stream>>>(Km, Qm, Vm, V1, gamma, out);
}
